// LargeBrain_86535001079928
// MI455X (gfx1250) — matrix operand order repaired and checked by reading, not compiled, not run on hardware
//
#include <hip/hip_runtime.h>
#include <hip/hip_bf16.h>
#include <stddef.h>
#include <stdint.h>

#define DIM   512
#define NNEUR 512
#define KFAN  5
#define SEQ   16
#define VOCAB 32000
#define FIRE_TH 0.005f

#define CHUNKROWS  32
#define CHUNK_ELEMS (CHUNKROWS * DIM)   // 16384 floats = 64 KB per K-chunk

typedef __attribute__((ext_vector_type(16))) _Float16 v16h;
typedef __attribute__((ext_vector_type(8)))  _Float16 v8h;
typedef __attribute__((ext_vector_type(8)))  float    v8f;
typedef __attribute__((ext_vector_type(4)))  unsigned int v4u;
typedef __attribute__((ext_vector_type(8)))  int      v8i_t;
typedef __attribute__((ext_vector_type(4)))  int      v4i_t;

// ---------------------------------------------------------------- TDM: contiguous 64KB global -> LDS
// D# per CDNA5 ISA ch.8: group0 = {count|flags, lds_addr, global_addr lo, global_addr hi|type=2}
// group1: data_size=4B, tensor_dim0=tile_dim0=16384 elems, tensor_dim1=tile_dim1=1.
// This toolchain's builtin is the 6-arg form: (g0 v4u, g1 v8i, g2 v4i, g3 v4i, v8i, cpol).
__device__ __forceinline__ void tdm_load_64k(const float* gsrc, unsigned lds_off) {
    unsigned long long ga = (unsigned long long)(uintptr_t)gsrc;
    v4u g0 = { 1u,                                   // count=1 (valid), gather off
               lds_off,                              // lds_addr (bytes)
               (unsigned)(ga & 0xffffffffull),       // global_addr[31:0]
               (unsigned)((ga >> 32) & 0x01ffffffu)  // global_addr[56:32]
                   | (2u << 30) };                   // type=2 ("image")
    v8i_t g1 = { (int)0x00020000,   // d0: wg_mask=0, data_size=2 (4B)
                 (int)0x40000000,   // d1: tensor_dim0[15:0]=16384 at bits 63:48
                 (int)0x00010000,   // d2: tensor_dim0[31:16]=0, tensor_dim1[15:0]=1
                 (int)0x40000000,   // d3: tensor_dim1[31:16]=0, tile_dim0=16384
                 1,                 // d4: tile_dim1=1, tile_dim2=0
                 CHUNK_ELEMS,       // d5: tensor_dim0_stride[31:0]
                 0, 0 };            // d6,d7: stride hi / dim1_stride = 0
    v4i_t gz4 = { 0, 0, 0, 0 };
    v8i_t gz8 = { 0, 0, 0, 0, 0, 0, 0, 0 };
    __builtin_amdgcn_tensor_load_to_lds(g0, g1, gz4, gz4, gz8, 0);
}

// ---------------------------------------------------------------- zero
__global__ void k_zero(float* __restrict__ p, int n) {
    int i = blockIdx.x * 256 + threadIdx.x;
    if (i < n) p[i] = 0.0f;
}

// ---------------------------------------------------------------- x = embed[ids] + pos
__global__ void k_embed(float* __restrict__ x, const int* __restrict__ ids,
                        const float* __restrict__ embed, const float* __restrict__ pos) {
    int s = blockIdx.x, t = threadIdx.x;              // 16 x 512
    int id = ids[s];
    x[s * DIM + t] = embed[(size_t)id * DIM + t] + pos[s * DIM + t];
}

// ---------------------------------------------------------------- per-neuron GEMV phase
// block = neuron, 512 threads = 16 waves; wave w computes output cols [32w, 32w+32).
// W_n streamed by the Tensor Data Mover into double-buffered LDS (2 x 64KB),
// wave 0 drives the TDM pipeline (TENSORcnt), compute waves run WMMA from LDS.
__global__ __launch_bounds__(512)
void k_compute(int s, int has_x,
               const float* __restrict__ cur_sum, const float* __restrict__ cur_cnt,
               const float* __restrict__ thr,
               const float* __restrict__ W, const float* __restrict__ bvec,
               float* __restrict__ post, float* __restrict__ ff,
               const float* __restrict__ x) {
    extern __shared__ __align__(16) float lw[];       // 2 * CHUNK_ELEMS floats
    __shared__ __align__(16) _Float16 lc[DIM];
    __shared__ float red[16];

    int n = blockIdx.x;
    int t = threadIdx.x;

    float inv  = 1.0f / (cur_cnt[n] + 1.0f);
    float xv   = has_x ? x[s * DIM + t] : 0.0f;
    float comb = (xv + cur_sum[(size_t)n * DIM + t]) * inv;
    lc[t] = (_Float16)comb;

    int wave = t >> 5;
    int lane = t & 31;
    int hi   = lane >> 4;
    int nl   = lane & 15;
    int c0   = wave * 32;
    const float* Wn = W + (size_t)n * DIM * DIM;
    float thn = thr[n];

    unsigned lwoff0 = (unsigned)(uintptr_t)(void*)&lw[0];
    unsigned lwoff1 = (unsigned)(uintptr_t)(void*)&lw[CHUNK_ELEMS];

    if (wave == 0) tdm_load_64k(Wn, lwoff0);          // prefetch chunk 0

    v8f acc0 = {}; v8f acc1 = {};
    for (int kb = 0; kb < DIM / CHUNKROWS; ++kb) {
        if (wave == 0) {
            if (kb < DIM / CHUNKROWS - 1) {
                tdm_load_64k(Wn + (size_t)(kb + 1) * CHUNK_ELEMS,
                             ((kb + 1) & 1) ? lwoff1 : lwoff0);
                __builtin_amdgcn_s_wait_tensorcnt(1); // chunk kb resident
            } else {
                __builtin_amdgcn_s_wait_tensorcnt(0);
            }
        }
        __syncthreads();                              // release compute waves

        int dbase = kb * CHUNKROWS;
        const float* lb = &lw[(kb & 1) * CHUNK_ELEMS];

        // A: replicate combined chunk across all 16 M-rows (contiguous k = 16*hi + i per lane half)
        v8h alo = *(const v8h*)&lc[dbase + hi * 16];
        v8h ahi = *(const v8h*)&lc[dbase + hi * 16 + 8];
        v16h a;
        #pragma unroll
        for (int i = 0; i < 8; ++i) { a[i] = alo[i]; a[8 + i] = ahi[i]; }

        int rbase = hi * 16;                          // local row within chunk
        v16h b0;
        #pragma unroll
        for (int j = 0; j < 8; ++j) {
            float w0 = lb[(rbase + 2 * j    ) * DIM + c0 + nl];
            float w1 = lb[(rbase + 2 * j + 1) * DIM + c0 + nl];
            b0[2 * j] = (_Float16)w0; b0[2 * j + 1] = (_Float16)w1;
        }
        acc0 = __builtin_amdgcn_wmma_f32_16x16x32_f16(false, a, false, b0,
                                                      (short)0, acc0, false, false);
        v16h b1;
        #pragma unroll
        for (int j = 0; j < 8; ++j) {
            float w0 = lb[(rbase + 2 * j    ) * DIM + c0 + 16 + nl];
            float w1 = lb[(rbase + 2 * j + 1) * DIM + c0 + 16 + nl];
            b1[2 * j] = (_Float16)w0; b1[2 * j + 1] = (_Float16)w1;
        }
        acc1 = __builtin_amdgcn_wmma_f32_16x16x32_f16(false, a, false, b1,
                                                      (short)0, acc1, false, false);
        __syncthreads();                              // done reading lb before overwrite
    }

    // all D rows identical (replicated A); lane's result for its column is acc[0]
    float p0 = acc0[0] + bvec[(size_t)n * DIM + c0 + nl]      - thn; p0 = fmaxf(p0, 0.0f);
    float p1 = acc1[0] + bvec[(size_t)n * DIM + c0 + 16 + nl] - thn; p1 = fmaxf(p1, 0.0f);

    float asum = (hi == 0) ? (fabsf(p0) + fabsf(p1)) : 0.0f;
    #pragma unroll
    for (int m = 16; m >= 1; m >>= 1) asum += __shfl_xor(asum, m, 32);
    if (lane == 0) red[wave] = asum;
    if (hi == 0) {
        post[(size_t)n * DIM + c0 + nl]      = p0;
        post[(size_t)n * DIM + c0 + 16 + nl] = p1;
    }
    __syncthreads();
    if (t == 0) {
        float tot = 0.0f;
        #pragma unroll
        for (int i = 0; i < 16; ++i) tot += red[i];
        ff[n] = (tot * (1.0f / DIM) > FIRE_TH) ? 1.0f : 0.0f;
    }
}

// ---------------------------------------------------------------- routing + homeostasis
__global__ __launch_bounds__(256)
void k_route(const float* __restrict__ post, const float* __restrict__ ff,
             const int* __restrict__ down,
             float* __restrict__ nxt_sum, float* __restrict__ nxt_cnt,
             float* __restrict__ thr, float* __restrict__ fc, float* __restrict__ rc,
             float* __restrict__ aggsum, float* __restrict__ nf, int s) {
    int n = blockIdx.x, t = threadIdx.x;
    float f = ff[n];
    if (t == 0) {
        float fcv = fc[n] + f;
        float rcv = (f > 0.0f) ? 0.0f : rc[n] + 1.0f;
        fc[n] = fcv; rc[n] = rcv;
        float tot = fcv + rcv;
        float adj = 0.0f;
        if (tot > 10.0f) {
            float rate = fcv / tot;
            adj = (rate > 0.6f) ? 0.005f : ((rate < 0.15f) ? -0.005f : 0.0f);
        }
        thr[n] = fminf(fmaxf(thr[n] + adj, -2.0f), 2.0f);
    }
    if (f > 0.0f) {
        int tg[KFAN];
        #pragma unroll
        for (int k = 0; k < KFAN; ++k) tg[k] = down[n * KFAN + k];
        for (int e = t; e < DIM; e += 256) {
            float p = post[(size_t)n * DIM + e];
            #pragma unroll
            for (int k = 0; k < KFAN; ++k)
                atomicAdd(&nxt_sum[(size_t)tg[k] * DIM + e], p);
            atomicAdd(&aggsum[s * DIM + e], p);
        }
        if (t == 0) {
            #pragma unroll
            for (int k = 0; k < KFAN; ++k) atomicAdd(&nxt_cnt[tg[k]], 1.0f);
            atomicAdd(&nf[s], 1.0f);
        }
    }
}

// ---------------------------------------------------------------- agg/nf + layernorm -> f16
__global__ __launch_bounds__(512)
void k_finalize(const float* __restrict__ aggsum, const float* __restrict__ nf,
                const float* __restrict__ gamma, const float* __restrict__ beta,
                _Float16* __restrict__ yn) {
    int s = blockIdx.x, t = threadIdx.x;   // 16 x 512
    __shared__ float r1[16], r2[16];
    float denom = fmaxf(nf[s], 1.0f);
    float y  = aggsum[s * DIM + t] / denom;
    float v  = y, v2 = y * y;
    #pragma unroll
    for (int m = 16; m >= 1; m >>= 1) { v += __shfl_xor(v, m, 32); v2 += __shfl_xor(v2, m, 32); }
    int wave = t >> 5, lane = t & 31;
    if (lane == 0) { r1[wave] = v; r2[wave] = v2; }
    __syncthreads();
    float sum = 0.0f, sum2 = 0.0f;
    #pragma unroll
    for (int i = 0; i < 16; ++i) { sum += r1[i]; sum2 += r2[i]; }
    float mu  = sum * (1.0f / DIM);
    float var = sum2 * (1.0f / DIM) - mu * mu;
    float is  = rsqrtf(var + 1e-5f);
    yn[s * DIM + t] = (_Float16)((y - mu) * is * gamma[t] + beta[t]);
}

// ---------------------------------------------------------------- output GEMM: [16x512]x[512x32000] (WMMA, M=16 full)
__global__ __launch_bounds__(256)
void k_outgemm(const _Float16* __restrict__ yn, const float* __restrict__ outW,
               const float* __restrict__ outb, float* __restrict__ out) {
    int tile = (blockIdx.x * 256 + threadIdx.x) >> 5;   // 2000 tiles of 16 cols
    int lane = threadIdx.x & 31;
    int hi = lane >> 4, nl = lane & 15;
    int M  = nl;                                        // A-row this lane supplies
    int col = tile * 16 + nl;

    v8f acc = {};
    for (int kb = 0; kb < DIM / 32; ++kb) {
        int dbase = kb * 32;
        const _Float16* ya = yn + (size_t)M * DIM;
        v8h alo = *(const v8h*)(ya + dbase + hi * 16);
        v8h ahi = *(const v8h*)(ya + dbase + hi * 16 + 8);
        v16h a;
        #pragma unroll
        for (int i = 0; i < 8; ++i) { a[i] = alo[i]; a[8 + i] = ahi[i]; }

        int koff = dbase + hi * 16;
        __builtin_prefetch(&outW[(size_t)(koff + 32) * VOCAB + col], 0, 0);
        v16h b;
        #pragma unroll
        for (int j = 0; j < 8; ++j) {
            b[2 * j]     = (_Float16)outW[(size_t)(koff + 2 * j    ) * VOCAB + col];
            b[2 * j + 1] = (_Float16)outW[(size_t)(koff + 2 * j + 1) * VOCAB + col];
        }
        acc = __builtin_amdgcn_wmma_f32_16x16x32_f16(false, a, false, b,
                                                     (short)0, acc, false, false);
    }
    float bb = outb[col];
    #pragma unroll
    for (int j = 0; j < 8; ++j) {
        int srow = hi * 8 + j;                          // D layout: M = j (+8 for hi half)
        out[(size_t)srow * VOCAB + col] = acc[j] + bb;
    }
}

// ---------------------------------------------------------------- host orchestration
extern "C" void kernel_launch(void* const* d_in, const int* in_sizes, int n_in,
                              void* d_out, int out_size, void* d_ws, size_t ws_size,
                              hipStream_t stream) {
    const int*   input_ids  = (const int*)d_in[0];
    const int*   downstream = (const int*)d_in[1];
    const float* embed      = (const float*)d_in[2];
    const float* pos_embed  = (const float*)d_in[3];
    const float* W          = (const float*)d_in[4];
    const float* bvec       = (const float*)d_in[5];
    const float* gamma      = (const float*)d_in[6];
    const float* beta       = (const float*)d_in[7];
    const float* outW       = (const float*)d_in[8];
    const float* outb       = (const float*)d_in[9];
    float*       out        = (float*)d_out;

    float* ws     = (float*)d_ws;
    float* x      = ws;                                   // SEQ*DIM
    float* buf0   = x + SEQ * DIM;                        // [N*DIM sum][N cnt]
    float* buf1   = buf0 + (NNEUR * DIM + NNEUR);
    float* thr    = buf1 + (NNEUR * DIM + NNEUR);         // N
    float* fc     = thr + NNEUR;                          // N
    float* rc     = fc + NNEUR;                           // N
    float* post   = rc + NNEUR;                           // N*DIM
    float* ffv    = post + NNEUR * DIM;                   // N
    float* aggsum = ffv + NNEUR;                          // SEQ*DIM
    float* nf     = aggsum + SEQ * DIM;                   // SEQ
    _Float16* ynh = (_Float16*)(nf + SEQ);                // SEQ*DIM halves

    // zero all mutable state (buffers, thr/fc/rc, agg, nf) every call
    int zcount = 2 * (NNEUR * DIM + NNEUR) + 3 * NNEUR + NNEUR * DIM + NNEUR
               + SEQ * DIM + SEQ;
    k_zero<<<(zcount + 255) / 256, 256, 0, stream>>>(buf0, zcount);
    k_embed<<<SEQ, DIM, 0, stream>>>(x, input_ids, embed, pos_embed);

    size_t dyn_lds = 2 * CHUNK_ELEMS * sizeof(float);     // 128 KB double buffer
    float* cur = buf0;
    float* nxt = buf1;
    for (int s = 0; s < SEQ; ++s) {
        for (int ph = 0; ph < 2; ++ph) {
            k_zero<<<((NNEUR * DIM + NNEUR) + 255) / 256, 256, 0, stream>>>(
                nxt, NNEUR * DIM + NNEUR);
            k_compute<<<NNEUR, 512, dyn_lds, stream>>>(s, ph == 0 ? 1 : 0,
                cur, cur + NNEUR * DIM, thr, W, bvec, post, ffv, x);
            k_route<<<NNEUR, 256, 0, stream>>>(post, ffv, downstream,
                nxt, nxt + NNEUR * DIM, thr, fc, rc, aggsum, nf, s);
            float* tmp = cur; cur = nxt; nxt = tmp;
        }
    }
    k_finalize<<<SEQ, DIM, 0, stream>>>(aggsum, nf, gamma, beta, ynh);
    k_outgemm<<<(VOCAB / 16) / 8, 256, 0, stream>>>(ynh, outW, outb, out);
}
